// CrossAttentionFusion_5686536700323
// MI455X (gfx1250) — compile-verified, matrix operand order checked by reading, not run on hardware
//
#include <hip/hip_runtime.h>
#include <hip/hip_bf16.h>

typedef _Float16 half_t;
typedef __attribute__((ext_vector_type(16))) _Float16 v16h;
typedef __attribute__((ext_vector_type(8)))  float    v8f;

#define Bc   4
#define Cc   256
#define Nn   4096
#define NT   256   // N/16 tiles
#define KC   8     // C/32 chunks
#define MC   128   // N/32 chunks
#define FRAG 512   // halves per 16x32 (or 32x16) fragment: 32 lanes * 16 halves

// ---- WMMA fragment index helpers (ISA 7.12.2, 16-bit A 16x32 / B 32x16) ----
// element (idx, k): lane = idx | ((k&8)<<1) ; e = (k&7) | ((k&16)>>1)
__device__ __forceinline__ int fragAB(int idx, int k) {
    return ((idx | ((k & 8) << 1)) << 4) | ((k & 7) | ((k & 16) >> 1));
}
// inverse: given (lane, e) -> k
__device__ __forceinline__ int invK(int lane, int e) {
    return (e & 7) | (((lane >> 4) & 1) << 3) | ((e & 8) << 1);
}

#define WMMA_F16(A, B, C) \
    __builtin_amdgcn_wmma_f32_16x16x32_f16(false, (A), false, (B), (short)0, (C), false, false)

// CDNA5 async global->LDS copy (GLOBAL_LOAD_ASYNC_TO_LDS_B128, ASYNCcnt-tracked)
__device__ __forceinline__ void async_ld16(unsigned lds_addr, unsigned long long gaddr) {
    asm volatile("global_load_async_to_lds_b128 %0, %1, off"
                 :: "v"(lds_addr), "v"(gaddr) : "memory");
}
__device__ __forceinline__ void wait_async0() {
    asm volatile("s_wait_asynccnt 0x0" ::: "memory");
}

// ---------------- prep: weights [256x256] f32 -> A-frag f16 ----------------
__global__ void prep_w_kernel(const float* __restrict__ w, half_t* __restrict__ dst) {
    int t = blockIdx.x * blockDim.x + threadIdx.x;   // 16*8*512 = 65536
    int e = t & 15, lane = (t >> 4) & 31, f = t >> 9;
    int kc = f & 7, mt = f >> 3;
    int m = lane & 15, k = invK(lane, e);
    int o = mt * 16 + m, c = kc * 32 + k;
    dst[t] = (half_t)w[o * Cc + c];
}

// ---------------- prep: x [B,C,N] f32 -> B-frag f16 ------------------------
__global__ void prep_x_kernel(const float* __restrict__ x, half_t* __restrict__ dst) {
    long t = (long)blockIdx.x * blockDim.x + threadIdx.x;  // B * 1048576
    int e = (int)(t & 15), lane = (int)((t >> 4) & 31);
    long f = t >> 9;
    int kc = (int)(f & 7), nt = (int)((f >> 3) & 255), b = (int)(f >> 11);
    int k = invK(lane, e);
    int c = kc * 32 + k, n = nt * 16 + (lane & 15);
    dst[t] = (half_t)x[((long)b * Cc + c) * Nn + n];
}

// -------- Q/K/V projection: W(A-frag) x X(B-frag) -> swizzled f16 ----------
// mode 0: Q  -> A-frag keyed (n-tile, c-chunk)   A[n][c] = q[c,n]
// mode 1: K  -> B-frag keyed (m-tile, c-chunk)   B[c][m] = k[c,m]
// mode 2: V  -> B-frag keyed (c-tile, m-chunk)   B[m][c] = v[c,m]
__global__ void qkv_kernel(const half_t* __restrict__ wfrag,
                           const half_t* __restrict__ xfrag,
                           const float*  __restrict__ bias,
                           half_t* __restrict__ dst, int mode) {
    int lane = threadIdx.x & 31;
    int gw = blockIdx.x * (blockDim.x >> 5) + (threadIdx.x >> 5);
    int nt = gw & 255, mt = (gw >> 8) & 15, b = gw >> 12;
    const long pbatch = (long)NT * KC * FRAG;
    const half_t* xb = xfrag + (long)b * pbatch;
    v8f d = {};
#pragma unroll
    for (int kc = 0; kc < 8; ++kc) {
        v16h a  = *(const v16h*)(wfrag + (mt * 8 + kc) * FRAG + lane * 16);
        v16h bb = *(const v16h*)(xb    + (nt * 8 + kc) * FRAG + lane * 16);
        d = WMMA_F16(a, bb, d);
    }
    half_t* db = dst + (long)b * pbatch;
    int h = (lane >> 4) & 1, ncol = lane & 15;
#pragma unroll
    for (int r = 0; r < 8; ++r) {
        int o = mt * 16 + r + 8 * h;     // output channel (D-layout M)
        int n = nt * 16 + ncol;          // spatial index  (D-layout N)
        float v = d[r] + bias[o];
        if (mode == 2) {  // V: B[k=m(n), col=c(o)]
            db[((long)((o >> 4) * MC + (n >> 5)) * FRAG) + fragAB(o & 15, n & 31)] = (half_t)v;
        } else {          // Q(A) / K(B): same (idx,k) mapping, idx=n, k=o
            db[((long)((n >> 4) * KC + (o >> 5)) * FRAG) + fragAB(n & 15, o & 31)] = (half_t)v;
        }
    }
}

// -------------------- flash attention --------------------------------------
// Block = 8 waves handling 8 query tiles of ONE batch; the 32KB K/V step is
// async-copied to LDS once per block (double-buffered), killing the 8x
// redundant L2 traffic. Q fragments live in registers for the whole kernel.
// LDS (dynamic, 72KB): bufK[2][16KB] | bufV[2][16KB] | P[8][1KB]
__global__ void __launch_bounds__(256, 1)
attn_kernel(const half_t* __restrict__ qf, const half_t* __restrict__ kf,
            const half_t* __restrict__ vf, half_t* __restrict__ ff) {
    extern __shared__ half_t smem[];
    const int lane = threadIdx.x & 31, wib = threadIdx.x >> 5, tid = threadIdx.x;
    const int b = blockIdx.x >> 5;               // 32 blocks per batch
    const int nt = ((blockIdx.x & 31) << 3) + wib;
    const long pbatch = (long)NT * KC * FRAG;
    const half_t* qb = qf + (long)b * pbatch;
    const half_t* kb = kf + (long)b * pbatch;
    const half_t* vb = vf + (long)b * pbatch;
    half_t* fb = ff + (long)b * pbatch;

    const unsigned ldsbase = (unsigned)(uintptr_t)smem;   // LDS byte offset (addr[31:0])
    const unsigned ldsK = ldsbase;                        // 2 x 16KB
    const unsigned ldsV = ldsbase + 32768;                // 2 x 16KB
    half_t* pl = smem + 32768 + wib * FRAG;               // per-wave P tile

    // resident Q fragments (8 x v16h = 64 VGPRs)
    v16h qa[8];
#pragma unroll
    for (int kc = 0; kc < 8; ++kc)
        qa[kc] = *(const v16h*)(qb + (nt * 8 + kc) * FRAG + lane * 16);

    v8f acc[16];
#pragma unroll
    for (int ct = 0; ct < 16; ++ct) acc[ct] = (v8f){};
    float mrun[8], lrun[8];
#pragma unroll
    for (int r = 0; r < 8; ++r) { mrun[r] = -1e30f; lrun[r] = 0.0f; }
    const int h = lane >> 4, ncol = lane & 15;
    const float scale = 0.0625f;  // C^-0.5

    // stage K/V step jn into buffer `buf` (256 threads x 8 x b128 = 32KB)
    auto issue = [&](int jn, int buf) {
        unsigned long long kg = (unsigned long long)kb + (unsigned long long)jn * 16384ull;
        unsigned kl = ldsK + buf * 16384;
#pragma unroll
        for (int i = 0; i < 4; ++i) {
            int off = (i * 256 + tid) * 16;
            async_ld16(kl + off, kg + off);
        }
        unsigned vl = ldsV + buf * 16384;
#pragma unroll
        for (int i = 0; i < 4; ++i) {
            int chunk = i * 256 + tid;            // 1024 x 16B
            int ct = chunk >> 6, within = (chunk & 63) << 4;
            async_ld16(vl + ct * 1024 + within,
                       (unsigned long long)vb + ((unsigned long long)(ct * MC + jn)) * 1024ull + within);
        }
    };

    issue(0, 0);
    for (int j = 0; j < MC; ++j) {
        const int cur = j & 1;
        wait_async0();            // own async copies for buf[cur] done
        __syncthreads();          // everyone's copies visible; prev buf free
        if (j + 1 < MC) issue(j + 1, cur ^ 1);

        const half_t* lk = smem + cur * 8192;            // 16 K frags
        const half_t* lv = smem + 16384 / 1 * 1 + cur * 8192 + 8192 * 0 + 16384; // see below
        lv = smem + 16384 + cur * 8192;                  // 16 V frags

        // S = Q'K'^T : two 16x16 tiles over 8 c-chunks
        v8f s0 = {}, s1 = {};
#pragma unroll
        for (int kc = 0; kc < 8; ++kc) {
            v16h k0 = *(const v16h*)(lk + kc * FRAG + lane * 16);
            v16h k1 = *(const v16h*)(lk + (8 + kc) * FRAG + lane * 16);
            s0 = WMMA_F16(qa[kc], k0, s0);
            s1 = WMMA_F16(qa[kc], k1, s1);
        }
        // online softmax: a row lives on 16 lanes of one half, per D-VGPR r
        float resc[8], p0[8], p1[8];
#pragma unroll
        for (int r = 0; r < 8; ++r) {
            float t0 = s0[r] * scale, t1 = s1[r] * scale;
            float mx = fmaxf(t0, t1);
            mx = fmaxf(mx, __shfl_xor(mx, 1, 32));
            mx = fmaxf(mx, __shfl_xor(mx, 2, 32));
            mx = fmaxf(mx, __shfl_xor(mx, 4, 32));
            mx = fmaxf(mx, __shfl_xor(mx, 8, 32));
            float mn = fmaxf(mrun[r], mx);
            float rs = __expf(mrun[r] - mn);
            mrun[r] = mn;
            float e0 = __expf(t0 - mn), e1 = __expf(t1 - mn);
            float sm = e0 + e1;
            sm += __shfl_xor(sm, 1, 32);
            sm += __shfl_xor(sm, 2, 32);
            sm += __shfl_xor(sm, 4, 32);
            sm += __shfl_xor(sm, 8, 32);
            lrun[r] = lrun[r] * rs + sm;
            resc[r] = rs; p0[r] = e0; p1[r] = e1;
        }
#pragma unroll
        for (int ct = 0; ct < 16; ++ct)
#pragma unroll
            for (int r = 0; r < 8; ++r) acc[ct][r] *= resc[r];
        // P: D-layout -> A-frag layout via per-wave LDS tile
#pragma unroll
        for (int r = 0; r < 8; ++r) {
            int n = r + 8 * h;
            pl[fragAB(n, ncol)]      = (half_t)p0[r];
            pl[fragAB(n, 16 + ncol)] = (half_t)p1[r];
        }
        asm volatile("s_wait_dscnt 0x0" ::: "memory");
        v16h pa = *(const v16h*)(pl + lane * 16);
        // F^T += P x V' over all 16 c-tiles
#pragma unroll
        for (int ct = 0; ct < 16; ++ct) {
            v16h vv = *(const v16h*)(lv + ct * FRAG + lane * 16);
            acc[ct] = WMMA_F16(pa, vv, acc[ct]);
        }
    }
    float inv[8];
#pragma unroll
    for (int r = 0; r < 8; ++r) inv[r] = 1.0f / lrun[r];
    // store fusion as B-frags for the proj GEMM: B[k=c][col=n]
#pragma unroll
    for (int ct = 0; ct < 16; ++ct)
#pragma unroll
        for (int r = 0; r < 8; ++r) {
            int n = r + 8 * h;
            int c = ct * 16 + ncol;
            fb[((long)(nt * 8 + (c >> 5)) * FRAG) + fragAB(n, c & 31)] =
                (half_t)(acc[ct][r] * inv[r]);
        }
}

// ---------------- gate: sigmoid(conv1x1(concat(x1,x2))) --------------------
__global__ void gate_kernel(const float* __restrict__ x1, const float* __restrict__ x2,
                            const float* __restrict__ gw, const float* __restrict__ gb,
                            float* __restrict__ gate) {
    int t = blockIdx.x * blockDim.x + threadIdx.x;  // B*N = 16384
    int n = t & (Nn - 1), b = t >> 12;
    const float* x1b = x1 + (long)b * Cc * Nn + n;
    const float* x2b = x2 + (long)b * Cc * Nn + n;
    float acc = gb[0];
    for (int c = 0; c < Cc; ++c)
        acc += gw[c] * x1b[(long)c * Nn] + gw[Cc + c] * x2b[(long)c * Nn];
    gate[t] = 1.0f / (1.0f + __expf(-acc));
}

// ------------- proj GEMM + BN(eval) + ReLU + gated residual ----------------
__global__ void proj_kernel(const half_t* __restrict__ wp, const half_t* __restrict__ ff,
                            const float* __restrict__ pb_, const float* __restrict__ gamma,
                            const float* __restrict__ beta, const float* __restrict__ mean,
                            const float* __restrict__ var,  const float* __restrict__ x1,
                            const float* __restrict__ gate, float* __restrict__ out) {
    int lane = threadIdx.x & 31;
    int gw = blockIdx.x * (blockDim.x >> 5) + (threadIdx.x >> 5);
    int nt = gw & 255, ot = (gw >> 8) & 15, b = gw >> 12;
    const half_t* fb = ff + (long)b * NT * KC * FRAG;
    v8f d = {};
#pragma unroll
    for (int kc = 0; kc < 8; ++kc) {
        v16h a  = *(const v16h*)(wp + (ot * 8 + kc) * FRAG + lane * 16);
        v16h bb = *(const v16h*)(fb + (nt * 8 + kc) * FRAG + lane * 16);
        d = WMMA_F16(a, bb, d);
    }
    int h = (lane >> 4) & 1, ncol = lane & 15;
    int n = nt * 16 + ncol;
    float g = gate[b * Nn + n];
#pragma unroll
    for (int r = 0; r < 8; ++r) {
        int o = ot * 16 + r + 8 * h;
        float y = d[r] + pb_[o];
        y = (y - mean[o]) * (gamma[o] * rsqrtf(var[o] + 1e-5f)) + beta[o];
        y = fmaxf(y, 0.0f);
        long idx = ((long)b * Cc + o) * Nn + n;
        out[idx] = x1[idx] + g * y;
    }
}

extern "C" void kernel_launch(void* const* d_in, const int* in_sizes, int n_in,
                              void* d_out, int out_size, void* d_ws, size_t ws_size,
                              hipStream_t stream) {
    const float* x1     = (const float*)d_in[0];
    const float* x2     = (const float*)d_in[1];
    const float* q_w    = (const float*)d_in[2];
    const float* q_b    = (const float*)d_in[3];
    const float* k_w    = (const float*)d_in[4];
    const float* k_b    = (const float*)d_in[5];
    const float* v_w    = (const float*)d_in[6];
    const float* v_b    = (const float*)d_in[7];
    const float* proj_w = (const float*)d_in[8];
    const float* proj_b = (const float*)d_in[9];
    const float* bn_g   = (const float*)d_in[10];
    const float* bn_b   = (const float*)d_in[11];
    const float* bn_m   = (const float*)d_in[12];
    const float* bn_v   = (const float*)d_in[13];
    const float* gate_w = (const float*)d_in[14];
    const float* gate_b = (const float*)d_in[15];
    float* out = (float*)d_out;

    char* ws = (char*)d_ws;
    const size_t WFRAG_B = 16 * 8 * 512 * sizeof(half_t);                // 128 KB
    const size_t XFRAG_B = (size_t)Bc * NT * KC * FRAG * sizeof(half_t); // 8 MB
    half_t* WQF = (half_t*)(ws);
    half_t* WKF = (half_t*)(ws + WFRAG_B);
    half_t* WVF = (half_t*)(ws + 2 * WFRAG_B);
    half_t* WPF = (half_t*)(ws + 3 * WFRAG_B);
    char* p = ws + 4 * WFRAG_B;
    half_t* X1F = (half_t*)p; p += XFRAG_B;
    half_t* X2F = (half_t*)p; p += XFRAG_B;
    half_t* QF  = (half_t*)p; p += XFRAG_B;
    half_t* KF  = (half_t*)p; p += XFRAG_B;
    half_t* VF  = (half_t*)p; p += XFRAG_B;
    half_t* FF  = (half_t*)p; p += XFRAG_B;
    float*  GATE = (float*)p;

    // 1. swizzle weights + activations to WMMA fragment layouts (f16)
    prep_w_kernel<<<256, 256, 0, stream>>>(q_w, WQF);
    prep_w_kernel<<<256, 256, 0, stream>>>(k_w, WKF);
    prep_w_kernel<<<256, 256, 0, stream>>>(v_w, WVF);
    prep_w_kernel<<<256, 256, 0, stream>>>(proj_w, WPF);
    prep_x_kernel<<<16384, 256, 0, stream>>>(x1, X1F);
    prep_x_kernel<<<16384, 256, 0, stream>>>(x2, X2F);

    // 2. Q/K/V projections (WMMA)
    qkv_kernel<<<2048, 256, 0, stream>>>(WQF, X1F, q_b, QF, 0);
    qkv_kernel<<<2048, 256, 0, stream>>>(WKF, X2F, k_b, KF, 1);
    qkv_kernel<<<2048, 256, 0, stream>>>(WVF, X2F, v_b, VF, 2);

    // 3. gate
    gate_kernel<<<64, 256, 0, stream>>>(x1, x2, gate_w, gate_b, GATE);

    // 4. flash attention (async LDS staging + WMMA QK^T + online softmax + WMMA PV)
    attn_kernel<<<128, 256, 73728, stream>>>(QF, KF, VF, FF);

    // 5. proj GEMM + BN + ReLU + gated residual
    proj_kernel<<<2048, 256, 0, stream>>>(WPF, FF, proj_b, bn_g, bn_b, bn_m, bn_v,
                                          x1, GATE, out);
}
